// ShallowDecoder_3221225472151
// MI455X (gfx1250) — matrix operand order repaired and checked by reading, not compiled, not run on hardware
//
#include <hip/hip_runtime.h>

#define DI __device__ __forceinline__

typedef __bf16 v16bf __attribute__((ext_vector_type(16)));
typedef __bf16 v8bf  __attribute__((ext_vector_type(8)));
typedef float  v8f   __attribute__((ext_vector_type(8)));

constexpr int KK = 77;          // text tokens
constexpr int BB = 256;         // batch
constexpr int DD = 768;         // d_model
constexpr int NN = 576;         // vision patches
constexpr int KP = 80;          // K padded to multiple of 16
constexpr float INV_T = 1.0f / 0.07f;

// pack 16 f32 -> v16bf (lowers to v_cvt_pk_bf16_f32 pairs)
DI v16bf cvt16(float4 x0, float4 x1, float4 x2, float4 x3) {
  v16bf r;
  r[0]  = (__bf16)x0.x; r[1]  = (__bf16)x0.y; r[2]  = (__bf16)x0.z; r[3]  = (__bf16)x0.w;
  r[4]  = (__bf16)x1.x; r[5]  = (__bf16)x1.y; r[6]  = (__bf16)x1.z; r[7]  = (__bf16)x1.w;
  r[8]  = (__bf16)x2.x; r[9]  = (__bf16)x2.y; r[10] = (__bf16)x2.z; r[11] = (__bf16)x2.w;
  r[12] = (__bf16)x3.x; r[13] = (__bf16)x3.y; r[14] = (__bf16)x3.z; r[15] = (__bf16)x3.w;
  return r;
}

// A-fragment (16x32 bf16) for GEMMs where A rows are contiguous f32 in memory.
// lane<16 -> K-subblocks {0..7,16..23}, lane>=16 -> {8..15,24..31}
DI v16bf load_a_frag(const float* row_d0_dsel) {
  const float4* p = (const float4*)(row_d0_dsel);
  const float4* q = (const float4*)(row_d0_dsel + 16);
  return cvt16(p[0], p[1], q[0], q[1]);
}

// ---------------------------------------------------------------------------
// Kernel 1: scores^T = vision_tile x text^T  (bf16 WMMA), exact softmax over n,
//           write attn as bf16 to workspace [b][KP][NN].
// grid = (BB, KP/16), block = 256 (8 waves)
// Each wave carries 4 independent C accumulators (n-tiles wave+8i, i=0..3),
// d-chunk loop outermost: 1 LDS B-fragment feeds 4 WMMA chains.
// ---------------------------------------------------------------------------
__global__ __launch_bounds__(256)
void scores_softmax_kernel(const float* __restrict__ text,
                           const float* __restrict__ vision,
                           __bf16* __restrict__ attn)
{
  __shared__ __align__(32) __bf16 tlds[DD * 16];   // text tile [k][d]  (24 KB)
  __shared__ float  slds[NN * 16];                 // scores^T   [n][k]  (36 KB)
  __shared__ float  rbuf[16 * 16];
  __shared__ float  rowmax[16];
  __shared__ float  rowsum[16];

  const int b  = blockIdx.x;
  const int k0 = blockIdx.y * 16;
  const int t  = threadIdx.x;

  // ---- stage text k-tile, transposed + converted to bf16 (zeros for k>=77)
  for (int idx = t; idx < 16 * DD; idx += 256) {
    const int k = idx / DD, d = idx % DD;
    float v = 0.0f;
    if (k0 + k < KK) v = text[((size_t)(k0 + k) * BB + b) * DD + d];
    tlds[k * DD + d] = (__bf16)v;
  }
  __syncthreads();

  const int wave = t >> 5;
  const int lane = t & 31;
  const int lmod = lane & 15;
  const int dsel = (lane < 16) ? 0 : 8;     // A-layout K-subblock select

  // ---- main pass: n-tiles 0..31, four per wave, d-chunk outermost
  {
    v8f c0 = {}, c1 = {}, c2 = {}, c3 = {};
    const float* r0 = vision + ((size_t)((wave +  0) * 16 + lmod) * BB + b) * DD + dsel;
    const float* r1 = vision + ((size_t)((wave +  8) * 16 + lmod) * BB + b) * DD + dsel;
    const float* r2 = vision + ((size_t)((wave + 16) * 16 + lmod) * BB + b) * DD + dsel;
    const float* r3 = vision + ((size_t)((wave + 24) * 16 + lmod) * BB + b) * DD + dsel;

    #pragma unroll 2
    for (int j = 0; j < DD / 32; ++j) {
      const int d0 = j * 32;
      // B fragment: text row lmod, d chunks {d0+dsel..+7, d0+dsel+16..+23} from LDS (loaded once)
      const v8bf b0 = *(const v8bf*)(tlds + (size_t)lmod * DD + d0 + dsel);
      const v8bf b1 = *(const v8bf*)(tlds + (size_t)lmod * DD + d0 + dsel + 16);
      const v16bf bm = __builtin_shufflevector(b0, b1, 0, 1, 2, 3, 4, 5, 6, 7,
                                               8, 9, 10, 11, 12, 13, 14, 15);
      // four independent accumulator chains share it
      c0 = __builtin_amdgcn_wmma_f32_16x16x32_bf16(false, load_a_frag(r0 + d0), false, bm,
                                                   (short)0, c0, false, false);
      c1 = __builtin_amdgcn_wmma_f32_16x16x32_bf16(false, load_a_frag(r1 + d0), false, bm,
                                                   (short)0, c1, false, false);
      c2 = __builtin_amdgcn_wmma_f32_16x16x32_bf16(false, load_a_frag(r2 + d0), false, bm,
                                                   (short)0, c2, false, false);
      c3 = __builtin_amdgcn_wmma_f32_16x16x32_bf16(false, load_a_frag(r3 + d0), false, bm,
                                                   (short)0, c3, false, false);
    }
    // C layout: VGPR r -> row n0 + r + 8*(lane/16), col k = lane%16
    const int nrow = (lane >> 4) * 8;
    #pragma unroll
    for (int r = 0; r < 8; ++r) {
      slds[(size_t)((wave +  0) * 16 + nrow + r) * 16 + lmod] = c0[r];
      slds[(size_t)((wave +  8) * 16 + nrow + r) * 16 + lmod] = c1[r];
      slds[(size_t)((wave + 16) * 16 + nrow + r) * 16 + lmod] = c2[r];
      slds[(size_t)((wave + 24) * 16 + nrow + r) * 16 + lmod] = c3[r];
    }
  }

  // ---- epilogue: n-tiles 32..35 handled by waves 0..3
  if (wave < 4) {
    const int n0 = (32 + wave) * 16;
    const float* r0 = vision + ((size_t)(n0 + lmod) * BB + b) * DD + dsel;
    v8f c = {};
    #pragma unroll 2
    for (int j = 0; j < DD / 32; ++j) {
      const int d0 = j * 32;
      const v8bf b0 = *(const v8bf*)(tlds + (size_t)lmod * DD + d0 + dsel);
      const v8bf b1 = *(const v8bf*)(tlds + (size_t)lmod * DD + d0 + dsel + 16);
      const v16bf bm = __builtin_shufflevector(b0, b1, 0, 1, 2, 3, 4, 5, 6, 7,
                                               8, 9, 10, 11, 12, 13, 14, 15);
      c = __builtin_amdgcn_wmma_f32_16x16x32_bf16(false, load_a_frag(r0 + d0), false, bm,
                                                  (short)0, c, false, false);
    }
    const int nrow = (lane >> 4) * 8;
    #pragma unroll
    for (int r = 0; r < 8; ++r)
      slds[(size_t)(n0 + nrow + r) * 16 + lmod] = c[r];
  }
  __syncthreads();

  // ---- exact softmax over n (columns of scores^T), 16 partials per k
  const int k    = t & 15;
  const int part = t >> 4;          // 0..15

  float mx = -3.0e38f;
  for (int n = part; n < NN; n += 16) mx = fmaxf(mx, slds[n * 16 + k]);
  rbuf[part * 16 + k] = mx;
  __syncthreads();
  if (t < 16) {
    float m = rbuf[t];
    #pragma unroll
    for (int i = 1; i < 16; ++i) m = fmaxf(m, rbuf[i * 16 + t]);
    rowmax[t] = m;
  }
  __syncthreads();

  const float m = rowmax[k];
  float s = 0.0f;
  for (int n = part; n < NN; n += 16) {
    const float e = __expf((slds[n * 16 + k] - m) * INV_T);
    slds[n * 16 + k] = e;
    s += e;
  }
  rbuf[part * 16 + k] = s;
  __syncthreads();
  if (t < 16) {
    float acc = 0.0f;
    #pragma unroll
    for (int i = 0; i < 16; ++i) acc += rbuf[i * 16 + t];
    rowsum[t] = acc;
  }
  __syncthreads();

  // ---- normalize + write attn bf16, coalesced along n
  const int kw = t >> 4;            // row of this k-tile
  const int nb = t & 15;
  const float rs = 1.0f / rowsum[kw];
  __bf16* arow = attn + ((size_t)b * KP + k0 + kw) * NN;
  for (int j = 0; j < NN / 16; ++j) {
    const int n = nb + j * 16;
    arow[n] = (__bf16)(slds[n * 16 + kw] * rs);
  }
}

// ---------------------------------------------------------------------------
// Kernel 2: out_tile(16k x 16d) += attn(16k x 32n) x vision(32n x 16d)
// grid = (BB, KP/16), block = 256 (8 waves); each wave owns 6 d-tiles
// (6 independent accumulator chains per shared A fragment).
// ---------------------------------------------------------------------------
__global__ __launch_bounds__(256)
void decode_kernel(const float* __restrict__ vision,
                   const __bf16* __restrict__ attn,
                   float* __restrict__ out)
{
  const int b    = blockIdx.x;
  const int k0   = blockIdx.y * 16;
  const int wave = threadIdx.x >> 5;
  const int lane = threadIdx.x & 31;
  const int lmod = lane & 15;

  v8f acc[6] = {};

  // A rows of attn for this lane (A layout chunk select folded into pointer)
  const __bf16* arow =
      attn + ((size_t)b * KP + k0 + lmod) * NN + ((lane < 16) ? 0 : 8);

  for (int ns = 0; ns < NN / 32; ++ns) {
    const int n0 = ns * 32;
    // A fragment: two contiguous 16B bf16 chunks
    const v8bf a0 = *(const v8bf*)(arow + n0);
    const v8bf a1 = *(const v8bf*)(arow + n0 + 16);
    const v16bf a = __builtin_shufflevector(a0, a1, 0, 1, 2, 3, 4, 5, 6, 7,
                                            8, 9, 10, 11, 12, 13, 14, 15);
    // B: lane -> d-col lmod, elements -> vision rows n0+8*(lane/16)+{0..7,16..23}
    const float* vrow = vision + ((size_t)(n0 + ((lane < 16) ? 0 : 8)) * BB + b) * DD + lmod;
    if (ns + 1 < NN / 32)
      __builtin_prefetch(vision + ((size_t)(n0 + 32 + lane) * BB + b) * DD, 0, 1);

    #pragma unroll
    for (int dt = 0; dt < 6; ++dt) {
      const int d0 = (wave + dt * 8) * 16;
      v16bf bm;
      #pragma unroll
      for (int s = 0; s < 16; ++s)
        bm[s] = (__bf16)vrow[(size_t)(s + ((s >= 8) ? 8 : 0)) * BB * DD + d0];
      acc[dt] = __builtin_amdgcn_wmma_f32_16x16x32_bf16(false, a, false, bm,
                                                        (short)0, acc[dt],
                                                        false, false);
    }
  }

  // C layout: VGPR r -> k = k0 + r + 8*(lane/16), d-col = lane%16
  #pragma unroll
  for (int dt = 0; dt < 6; ++dt) {
    const int d0 = (wave + dt * 8) * 16;
    #pragma unroll
    for (int r = 0; r < 8; ++r) {
      const int kk = k0 + r + 8 * (lane >> 4);
      if (kk < KK)
        out[((size_t)kk * BB + b) * DD + d0 + lmod] = acc[dt][r];
    }
  }
}

// ---------------------------------------------------------------------------
extern "C" void kernel_launch(void* const* d_in, const int* in_sizes, int n_in,
                              void* d_out, int out_size, void* d_ws, size_t ws_size,
                              hipStream_t stream) {
  const float* text   = (const float*)d_in[0];   // [77][256][768] f32
  const float* vision = (const float*)d_in[1];   // [576][256][768] f32
  float* out  = (float*)d_out;                   // [77][256][768] f32
  __bf16* attn = (__bf16*)d_ws;                  // [256][80][576] bf16 (~23.6 MB)

  dim3 grid(BB, KP / 16);
  scores_softmax_kernel<<<grid, 256, 0, stream>>>(text, vision, attn);
  decode_kernel<<<grid, 256, 0, stream>>>(vision, attn, out);
}
